// Policy_43138651521637
// MI455X (gfx1250) — 8-bit matrix operand order repaired and checked by reading, compiled, not run on hardware
//
#include <hip/hip_runtime.h>

typedef __attribute__((ext_vector_type(8))) int v8i;

#define B_DIM 512
#define F_DIM 1180
#define A_DIM 50
#define C_DIM 1024
#define N_DIM (A_DIM * C_DIM)
#define KCH 64
#define NCHUNKS 19
#define NT 64
#define THREADS 256
#define MID 16.0f

__global__ void pack_x_kernel(const float* __restrict__ x,
                              unsigned int* __restrict__ xp) {
  int t = blockIdx.x * THREADS + threadIdx.x;
  int v    = t & 7;
  int lane = (t >> 3) & 31;
  int mt   = (t >> 8) & 31;
  int ch   = t >> 13;
  int m  = mt * 16 + (lane & 15);
  int hi = lane >> 4;
  int kin = ((v & 1) << 2) | (((v >> 1) & 1) << 3) | (hi << 4) | (((v >> 2) & 1) << 5);
  int k = ch * KCH + kin;
  unsigned int val = 0u;
  if (k < F_DIM) {
    float4 xf = *(const float4*)(x + (size_t)m * F_DIM + k);
    unsigned int b0 = xf.x != 0.f, b1 = xf.y != 0.f, b2 = xf.z != 0.f, b3 = xf.w != 0.f;
    val = b0 | (b1 << 8) | (b2 << 16) | (b3 << 24);
  }
  xp[t] = val;
}

__global__ void __launch_bounds__(THREADS, 1)
tm_gemm_kernel(const float* __restrict__ ta,
               const unsigned int* __restrict__ xp,
               const float* __restrict__ csign,
               int* __restrict__ votes)
{
  __shared__ __attribute__((aligned(32))) unsigned int wfrag[4 * 256];
  __shared__ int poscnt[NT];
  __shared__ int sgn[NT];

  const int tid  = threadIdx.x;
  const int lane = tid & 31;
  const int wv   = tid >> 5;
  const int n0   = blockIdx.x * NT;
  const int a_class = n0 / C_DIM;

  if (tid < NT) {
    poscnt[tid] = 0;
    sgn[tid] = (csign[n0 + tid] >= 0.f) ? 1 : -1;
  }
  __syncthreads();

  const v8i vzero = {0, 0, 0, 0, 0, 0, 0, 0};
  v8i acc[4][4];
#pragma unroll
  for (int i = 0; i < 4; ++i)
#pragma unroll
    for (int j = 0; j < 4; ++j) acc[i][j] = vzero;

  int pc[4] = {0, 0, 0, 0};

  for (int ch = 0; ch < NCHUNKS; ++ch) {
#pragma unroll
    for (int it = 0; it < 4; ++it) {
      int q  = tid + it * THREADS;
      int nl = q >> 4;
      int f4 = q & 15;
      int kg = ch * KCH + f4 * 4;
      unsigned int w32 = 0u;
      if (kg < F_DIM) {
        size_t base = ((size_t)(n0 + nl) * 2) * F_DIM + kg;
        float4 pos = *(const float4*)(ta + base);
        float4 neg = *(const float4*)(ta + base + F_DIM);
        int p0 = pos.x > MID, p1 = pos.y > MID, p2 = pos.z > MID, p3 = pos.w > MID;
        pc[it] += p0 + p1 + p2 + p3;
        int w0 = (int)(neg.x > MID) - p0;
        int w1 = (int)(neg.y > MID) - p1;
        int w2 = (int)(neg.z > MID) - p2;
        int w3 = (int)(neg.w > MID) - p3;
        w32 = (w0 & 0xff) | ((w1 & 0xff) << 8) | ((w2 & 0xff) << 16) | ((w3 & 0xff) << 24);
      }
      int lf = (nl & 15) + (((f4 >> 2) & 1) << 4);
      int vv = (f4 & 3) + ((f4 >> 3) << 2);
      int j  = nl >> 4;
      wfrag[j * 256 + lf * 8 + vv] = w32;
    }
    __syncthreads();

    v8i afr[4];
#pragma unroll
    for (int i = 0; i < 4; ++i) {
      int mt = wv * 4 + i;
      afr[i] = *(const v8i*)(xp + (((size_t)ch * 32 + mt) * 32 + lane) * 8);
    }
#pragma unroll
    for (int j = 0; j < 4; ++j) {
      v8i bfr = *(const v8i*)(&wfrag[j * 256 + lane * 8]);
#pragma unroll
      for (int i = 0; i < 4; ++i) {
        acc[i][j] = __builtin_amdgcn_wmma_i32_16x16x64_iu8(
            false, afr[i], true, bfr, acc[i][j], false, false);
      }
    }
    __syncthreads();
  }

#pragma unroll
  for (int it = 0; it < 4; ++it) {
    int q = tid + it * THREADS;
    atomicAdd(&poscnt[q >> 4], pc[it]);
  }
  __syncthreads();

  const int ncol = lane & 15;
  const int hi   = lane >> 4;
#pragma unroll
  for (int i = 0; i < 4; ++i) {
#pragma unroll
    for (int r = 0; r < 8; ++r) {
      int s = 0;
#pragma unroll
      for (int j = 0; j < 4; ++j) {
        int nc = j * 16 + ncol;
        int viol = poscnt[nc] + acc[i][j][r];
        s += (viol == 0) ? sgn[nc] : 0;
      }
      s += __shfl_xor(s, 1, 32);
      s += __shfl_xor(s, 2, 32);
      s += __shfl_xor(s, 4, 32);
      s += __shfl_xor(s, 8, 32);
      if (ncol == 0) {
        int m = (wv * 4 + i) * 16 + r + 8 * hi;
        atomicAdd(&votes[m * A_DIM + a_class], s);
      }
    }
  }
}

__global__ void finalize_kernel(const int* __restrict__ votes,
                                const int* __restrict__ Tptr,
                                float* __restrict__ out) {
  int i = blockIdx.x * THREADS + threadIdx.x;
  if (i < B_DIM * A_DIM) {
    float Tf = (float)Tptr[0];
    float v = (float)votes[i];
    out[i] = fminf(fmaxf(v, -Tf), Tf);
  }
}

extern "C" void kernel_launch(void* const* d_in, const int* in_sizes, int n_in,
                              void* d_out, int out_size, void* d_ws, size_t ws_size,
                              hipStream_t stream) {
  const float* x  = (const float*)d_in[0];
  const float* ta = (const float*)d_in[1];
  const float* cs = (const float*)d_in[2];
  const int*   Tp = (const int*)d_in[3];

  int* votes = (int*)d_ws;
  unsigned int* xpack = (unsigned int*)((char*)d_ws + (size_t)B_DIM * A_DIM * 4);

  hipMemsetAsync(votes, 0, (size_t)B_DIM * A_DIM * 4, stream);

  pack_x_kernel<<<(NCHUNKS * 32 * 32 * 8) / THREADS, THREADS, 0, stream>>>(x, xpack);
  tm_gemm_kernel<<<N_DIM / NT, THREADS, 0, stream>>>(ta, xpack, cs, votes);
  finalize_kernel<<<(B_DIM * A_DIM + THREADS - 1) / THREADS, THREADS, 0, stream>>>(
      votes, Tp, (float*)d_out);
}
